// ModifiedAttention_53120155517143
// MI455X (gfx1250) — matrix operand order repaired and checked by reading, not compiled, not run on hardware
//
#include <hip/hip_runtime.h>

// Problem constants (reference: H=2048, NH=16, HD=128, B=2, S=2048)
#define H_DIM 2048
#define NHEADS 16
#define HEAD_DIM 128
#define SEQ 2048
#define BATCH 2

typedef __attribute__((ext_vector_type(16))) __bf16 v16bf;
typedef __attribute__((ext_vector_type(8)))  float  v8f;
typedef __attribute__((ext_vector_type(4)))  unsigned u32x4;
typedef __attribute__((ext_vector_type(8)))  int      i32x8;
typedef __attribute__((ext_vector_type(4)))  int      i32x4;

#if defined(__has_builtin)
#if __has_builtin(__builtin_amdgcn_tensor_load_to_lds) && \
    __has_builtin(__builtin_amdgcn_s_wait_tensorcnt)
#define HAVE_TDM 1
#endif
#endif

// ---------------- helpers ----------------

__device__ inline __bf16 f32_to_bf16(float f) {
  unsigned u = __builtin_bit_cast(unsigned, f);
  u += 0x7FFFu + ((u >> 16) & 1u);          // round-to-nearest-even
  unsigned short h = (unsigned short)(u >> 16);
  return __builtin_bit_cast(__bf16, h);
}

// A-fragment, 16x32 bf16 (M x K), source row-major with stride lda.
// Layout: lanes 0-15 row M=lane, K 0..15; lanes 16-31 same rows, K 16..31.
__device__ inline v16bf load_a_frag(const __bf16* __restrict__ src, int lda) {
  const int lane = threadIdx.x & 31;
  const int row  = lane & 15;
  const int koff = (lane >> 4) << 4;        // 0 or 16
  const __bf16* p = src + (size_t)row * lda + koff;
  v16bf a;
#pragma unroll
  for (int i = 0; i < 16; ++i) a[i] = p[i];
  return a;
}

// B-fragment, 32x16 bf16 (K x N), operand held as [N][K] row-major (W, K^T, V^T).
// Contiguous k order (matches load_a_frag): lanes 0-15 -> N=lane, K=0..15; lanes 16-31 -> N=lane-16, K=16..31.
__device__ inline v16bf load_bT_frag(const __bf16* __restrict__ src, int ldb) {
  const int lane = threadIdx.x & 31;
  const int col  = lane & 15;
  const int koff = (lane >> 4) << 4;        // 0 or 16
  const __bf16* p = src + (size_t)col * ldb + koff;
  v16bf b;
#pragma unroll
  for (int j = 0; j < 16; ++j) b[j] = p[j];
  return b;
}

__device__ inline v8f wmma_bf16(v16bf a, v16bf b, v8f c) {
  return __builtin_amdgcn_wmma_f32_16x16x32_bf16(false, a, false, b, (short)0, c,
                                                 false, false);
}

#ifdef HAVE_TDM
// TDM 2D descriptor, split so the loop-invariant part (group1: dims/strides/tile
// shape, D# layout per CDNA5 ISA 8.3/8.4) is built ONCE and stays in SGPRs;
// per-issue we patch only the LDS address and global tile address in group0.
struct Tdm2D {
  i32x8 g1;
};

__device__ inline Tdm2D tdm_make_2d(unsigned dim0, unsigned dim1, unsigned stride0,
                                    unsigned tile0, unsigned tile1) {
  Tdm2D t;
  t.g1[0] = (int)(1u << 16);                                  // data_size=1 (2 bytes)
  t.g1[1] = (int)((dim0 & 0xFFFFu) << 16);                    // tensor_dim0[15:0]
  t.g1[2] = (int)((dim0 >> 16) | ((dim1 & 0xFFFFu) << 16));   // dim0 hi | dim1 lo
  t.g1[3] = (int)((dim1 >> 16) | (tile0 << 16));              // dim1 hi | tile_dim0
  t.g1[4] = (int)tile1;                                       // tile_dim1 (tile_dim2=0)
  t.g1[5] = (int)stride0;                                     // tensor_dim0_stride lo
  t.g1[6] = 0;
  t.g1[7] = 0;
  return t;
}

__device__ inline void tdm_issue(const Tdm2D& t, unsigned lds_addr, const void* g) {
  unsigned long long ga = (unsigned long long)(uintptr_t)g;
  u32x4 g0;
  g0[0] = 1u;                                                 // count=1, user mode
  g0[1] = lds_addr;                                           // LDS byte address
  g0[2] = (unsigned)ga;                                       // global_addr[31:0]
  g0[3] = (unsigned)((ga >> 32) & 0x1FFFFFFu) | (2u << 30);   // addr[56:32] | type=2
  i32x4 z4 = {0, 0, 0, 0};
#if __has_include(<hip/amd_detail/amd_gfx1250_TDM.h>)
  i32x8 z8 = {0, 0, 0, 0, 0, 0, 0, 0};                        // clang-23: 6-arg form
  __builtin_amdgcn_tensor_load_to_lds(g0, t.g1, z4, z4, z8, 0);
#else
  __builtin_amdgcn_tensor_load_to_lds(g0, t.g1, z4, z4, 0);   // ROCm 7.2: 5-arg form
#endif
}
#endif

// ---------------- fp32 -> bf16 cast ----------------

__global__ void cast_f32_bf16(const float* __restrict__ in,
                              __bf16* __restrict__ out, int n) {
  int i = blockIdx.x * blockDim.x + threadIdx.x;
  int stride = gridDim.x * blockDim.x;
  for (; i < n; i += stride) out[i] = f32_to_bf16(in[i]);
}

// ---------------- GEMM: out = A[M,K] @ W[N,K]^T + bias ----------------
// MODE 0: bf16 head-major [B][NH][S][HD] with post-scale (Q,K).
// MODE 1: fp32 row-major [M,N] (output projection).
// MODE 2: bf16 transposed head-major [B][NH][HD][S] (V, so PV B-operand is contiguous).
// W tile double-buffered in LDS; TDM DMA for tile i+1 overlaps compute on tile i.
template <int MODE>
__global__ void __launch_bounds__(256)
gemm_x_wT(const __bf16* __restrict__ A, const __bf16* __restrict__ W,
          const float* __restrict__ bias, void* __restrict__ out,
          int M, int N, int K, float scale) {
  __shared__ __bf16 lds_b[2][128 * 32];    // 2 x (128 N-cols x 32 K) = 16 KB

  const int wave  = threadIdx.x >> 5;
  const int lane  = threadIdx.x & 31;
  const int col   = lane & 15;
  const int hi    = lane >> 4;
  const int mbase = blockIdx.x * 128 + wave * 16;   // each wave: 16 rows
  const int nbase = blockIdx.y * 128;

  v8f c[8];
#pragma unroll
  for (int f = 0; f < 8; ++f) {
    float bv = bias[nbase + f * 16 + col];
#pragma unroll
    for (int r = 0; r < 8; ++r) c[f][r] = bv;
  }

  const __bf16* Arow  = A + (size_t)mbase * K;
  const __bf16* Wtile = W + (size_t)nbase * K;

#ifdef HAVE_TDM
  const Tdm2D dW = tdm_make_2d((unsigned)K, (unsigned)N, (unsigned)K, 32u, 128u);
  const unsigned lds_ba[2] = { (unsigned)(uintptr_t)&lds_b[0][0],
                               (unsigned)(uintptr_t)&lds_b[1][0] };
  if (wave == 0)               // prologue: DMA tile 0 into buffer 0
    tdm_issue(dW, lds_ba[0], Wtile);
#endif

  for (int kb = 0, it = 0; kb < K; kb += 32, ++it) {
    const int cur = it & 1;
    __syncthreads();           // all waves done reading the buffer about to be overwritten
#ifdef HAVE_TDM
    if (wave == 0) {
      if (kb + 32 < K) {       // DMA next tile into the other buffer
        tdm_issue(dW, lds_ba[cur ^ 1], Wtile + kb + 32);
        __builtin_amdgcn_s_wait_tensorcnt(1);   // in-order: current tile has landed
      } else {
        __builtin_amdgcn_s_wait_tensorcnt(0);
      }
    }
#else
    {  // cooperative stage of W tile: 128 cols x 32 k as [col][k]
      int ccol = threadIdx.x >> 1;
      int half = threadIdx.x & 1;
      const uint4* src = (const uint4*)(Wtile + (size_t)ccol * K + kb + half * 16);
      uint4* dst = (uint4*)(&lds_b[cur][ccol * 32 + half * 16]);
      dst[0] = src[0];
      dst[1] = src[1];
    }
#endif
    if (kb + 32 < K) __builtin_prefetch(Arow + kb + 32, 0, 3);  // global_prefetch
    __syncthreads();

    v16bf a = load_a_frag(Arow + kb, K);
    v16bf bfr[8];
#pragma unroll
    for (int f = 0; f < 8; ++f)
      bfr[f] = load_bT_frag(&lds_b[cur][f * 16 * 32], 32);
#pragma unroll
    for (int f = 0; f < 8; ++f)
      c[f] = wmma_bf16(a, bfr[f], c[f]);
  }

  if (MODE == 0) {
    __bf16* o = (__bf16*)out;   // [B][NH][S][HD]
#pragma unroll
    for (int f = 0; f < 8; ++f) {
      int n = nbase + f * 16 + col;
      int h = n >> 7, d = n & (HEAD_DIM - 1);
#pragma unroll
      for (int r = 0; r < 8; ++r) {
        int m = mbase + r + hi * 8;
        int b = m >> 11, s = m & (SEQ - 1);
        size_t idx = (((size_t)b * NHEADS + h) * SEQ + s) * HEAD_DIM + d;
        o[idx] = f32_to_bf16(c[f][r] * scale);
      }
    }
  } else if (MODE == 2) {
    __bf16* o = (__bf16*)out;   // [B][NH][HD][S]
#pragma unroll
    for (int f = 0; f < 8; ++f) {
      int n = nbase + f * 16 + col;
      int h = n >> 7, d = n & (HEAD_DIM - 1);
#pragma unroll
      for (int r = 0; r < 8; ++r) {
        int m = mbase + r + hi * 8;
        int b = m >> 11, s = m & (SEQ - 1);
        size_t idx = (((size_t)b * NHEADS + h) * HEAD_DIM + d) * SEQ + s;
        o[idx] = f32_to_bf16(c[f][r]);
      }
    }
  } else {
    float* o = (float*)out;     // [M][N]
#pragma unroll
    for (int f = 0; f < 8; ++f) {
      int n = nbase + f * 16 + col;
#pragma unroll
      for (int r = 0; r < 8; ++r) {
        int m = mbase + r + hi * 8;
        o[(size_t)m * N + n] = c[f][r];
      }
    }
  }
}

// ---------------- Flash attention (scale pre-folded into Q) ----------------
// grid: (SEQ/128, BATCH*NHEADS), block 256 = 8 waves; each wave: 16 Q rows.
// K / V^T tiles double-buffered in LDS; TDM DMA overlaps the previous tile's compute.
__global__ void __launch_bounds__(256)
flash_attn(const __bf16* __restrict__ Q, const __bf16* __restrict__ K,
           const __bf16* __restrict__ Vt, __bf16* __restrict__ O) {
  __shared__ __bf16 lds_k[2][32 * 128];    // [key][d]   2 x 8 KB
  __shared__ __bf16 lds_vt[2][128 * 32];   // [d][key]   2 x 8 KB
  __shared__ __bf16 lds_p[8][16 * 32];     // wave-private P staging (C->A relayout)

  const int wave = threadIdx.x >> 5;
  const int lane = threadIdx.x & 31;
  const int col  = lane & 15;
  const int hi   = lane >> 4;
  const int bh   = blockIdx.y;             // b*NH + h
  const int qbase = blockIdx.x * 128 + wave * 16;

  const size_t head = (size_t)bh * SEQ * HEAD_DIM;  // same extent for [S,HD] and [HD,S]
  const __bf16* Qh  = Q + head;
  const __bf16* Kh  = K + head;
  const __bf16* Vth = Vt + head;

  // Q fragments: 16 rows x 128 head-dim = 4 A-frags, loaded once.
  v16bf qf[4];
#pragma unroll
  for (int i = 0; i < 4; ++i)
    qf[i] = load_a_frag(Qh + (size_t)qbase * HEAD_DIM + i * 32, HEAD_DIM);

  // all-ones B operand: P x ones puts the row-sum in every column of C
  v16bf ones;
#pragma unroll
  for (int j = 0; j < 16; ++j)
    ones[j] = __builtin_bit_cast(__bf16, (unsigned short)0x3F80);

  v8f o[8];
#pragma unroll
  for (int f = 0; f < 8; ++f)
#pragma unroll
    for (int r = 0; r < 8; ++r) o[f][r] = 0.f;
  float mrow[8], lrow[8], corr[8];
#pragma unroll
  for (int r = 0; r < 8; ++r) { mrow[r] = -3.0e38f; lrow[r] = 0.f; }

#ifdef HAVE_TDM
  const Tdm2D dK = tdm_make_2d(HEAD_DIM, SEQ, HEAD_DIM, HEAD_DIM, 32u);
  const Tdm2D dV = tdm_make_2d(SEQ, HEAD_DIM, SEQ, 32u, HEAD_DIM);
  const unsigned lds_ka[2] = { (unsigned)(uintptr_t)&lds_k[0][0],
                               (unsigned)(uintptr_t)&lds_k[1][0] };
  const unsigned lds_va[2] = { (unsigned)(uintptr_t)&lds_vt[0][0],
                               (unsigned)(uintptr_t)&lds_vt[1][0] };
  if (wave == 0) {             // prologue: DMA tile 0 (K + V^T) into buffer 0
    tdm_issue(dK, lds_ka[0], Kh);
    tdm_issue(dV, lds_va[0], Vth);
  }
#endif

  for (int kb = 0, it = 0; kb < SEQ; kb += 32, ++it) {
    const int cur = it & 1;
    __syncthreads();           // reads of the buffer about to be overwritten are done
#ifdef HAVE_TDM
    if (wave == 0) {
      if (kb + 32 < SEQ) {     // DMA next K+V^T pair into the other buffer
        tdm_issue(dK, lds_ka[cur ^ 1], Kh + (size_t)(kb + 32) * HEAD_DIM);
        tdm_issue(dV, lds_va[cur ^ 1], Vth + kb + 32);
        __builtin_amdgcn_s_wait_tensorcnt(2);   // in-order: current pair has landed
      } else {
        __builtin_amdgcn_s_wait_tensorcnt(0);
      }
    }
#else
    {  // cooperative staging, 16 bf16 per thread per tile
      int t = threadIdx.x;
      int krow = t >> 3, kseg = t & 7;
      const uint4* sk = (const uint4*)(Kh + (size_t)(kb + krow) * HEAD_DIM + kseg * 16);
      uint4* dk = (uint4*)(&lds_k[cur][krow * HEAD_DIM + kseg * 16]);
      dk[0] = sk[0]; dk[1] = sk[1];
      int drow = t >> 1, dhalf = t & 1;
      const uint4* sv = (const uint4*)(Vth + (size_t)drow * SEQ + kb + dhalf * 16);
      uint4* dv = (uint4*)(&lds_vt[cur][drow * 32 + dhalf * 16]);
      dv[0] = sv[0]; dv[1] = sv[1];
    }
#endif
    __syncthreads();

    // scores for 32 keys: 8 K^T B-frags from LDS, two C accumulators interleaved
    v8f s0 = {}, s1 = {};
    {
      v16bf kb0[4], kb1[4];
#pragma unroll
      for (int i = 0; i < 4; ++i) {
        kb0[i] = load_bT_frag(&lds_k[cur][0 * HEAD_DIM + i * 32], HEAD_DIM);
        kb1[i] = load_bT_frag(&lds_k[cur][16 * HEAD_DIM + i * 32], HEAD_DIM);
      }
#pragma unroll
      for (int i = 0; i < 4; ++i) {
        s0 = wmma_bf16(qf[i], kb0[i], s0);
        s1 = wmma_bf16(qf[i], kb1[i], s1);
      }
    }

    // online softmax: butterfly for row-max only (row-sum comes from a WMMA)
#pragma unroll
    for (int r = 0; r < 8; ++r) {
      float mx = fmaxf(s0[r], s1[r]);
#pragma unroll
      for (int d = 1; d < 16; d <<= 1) mx = fmaxf(mx, __shfl_xor(mx, d, 32));
      float mnew = fmaxf(mrow[r], mx);
      corr[r] = __expf(mrow[r] - mnew);
      mrow[r] = mnew;
      float p0 = __expf(s0[r] - mnew);
      float p1 = __expf(s1[r] - mnew);
      int row = r + hi * 8;                 // stage P row-major 16x32 (wave-private)
      lds_p[wave][row * 32 + col]      = f32_to_bf16(p0);
      lds_p[wave][row * 32 + 16 + col] = f32_to_bf16(p1);
#pragma unroll
      for (int f = 0; f < 8; ++f) o[f][r] *= corr[r];
    }

    // P(16x32): row-sum via ones-WMMA, then PV accumulation (contiguous V^T B-frags)
    v16bf pf = load_a_frag(lds_p[wave], 32);
    v16bf vb[8];
#pragma unroll
    for (int f = 0; f < 8; ++f)
      vb[f] = load_bT_frag(&lds_vt[cur][(f * 16) * 32], 32);
    v8f rsum = {};
    rsum = wmma_bf16(pf, ones, rsum);
#pragma unroll
    for (int r = 0; r < 8; ++r) lrow[r] = lrow[r] * corr[r] + rsum[r];
#pragma unroll
    for (int f = 0; f < 8; ++f)
      o[f] = wmma_bf16(pf, vb[f], o[f]);
  }

  // normalize + write O as [B][S][H] bf16 (A-operand of the output projection)
  const int b = bh >> 4, h = bh & (NHEADS - 1);
#pragma unroll
  for (int r = 0; r < 8; ++r) {
    float inv = 1.f / lrow[r];
    int s = qbase + r + hi * 8;
    size_t rowoff = ((size_t)b * SEQ + s) * H_DIM + (size_t)h * HEAD_DIM;
#pragma unroll
    for (int f = 0; f < 8; ++f)
      O[rowoff + f * 16 + col] = f32_to_bf16(o[f][r] * inv);
  }
}

// ---------------- host orchestration ----------------

extern "C" void kernel_launch(void* const* d_in, const int* in_sizes, int n_in,
                              void* d_out, int out_size, void* d_ws, size_t ws_size,
                              hipStream_t stream) {
  const float* hs = (const float*)d_in[0];
  const float* Wq = (const float*)d_in[1];
  const float* bq = (const float*)d_in[2];
  const float* Wk = (const float*)d_in[3];
  const float* bk = (const float*)d_in[4];
  const float* Wv = (const float*)d_in[5];
  const float* bv = (const float*)d_in[6];
  const float* Wo = (const float*)d_in[7];
  const float* bo = (const float*)d_in[8];

  const size_t M  = (size_t)BATCH * SEQ;        // 4096
  const size_t HH = (size_t)H_DIM * H_DIM;      // 2048*2048
  const size_t XN = M * H_DIM;                  // activation elems

  char* ws = (char*)d_ws;
  size_t off = 0;
  __bf16* Xbf  = (__bf16*)(ws + off); off += XN * 2;   // 16 MB (reused as Obf)
  __bf16* Wbf0 = (__bf16*)(ws + off); off += HH * 2;   // 8 MB  (Wq, reused for Wo)
  __bf16* Wbf1 = (__bf16*)(ws + off); off += HH * 2;   // Wk
  __bf16* Wbf2 = (__bf16*)(ws + off); off += HH * 2;   // Wv
  __bf16* Qbf  = (__bf16*)(ws + off); off += XN * 2;   // [B,NH,S,HD]
  __bf16* Kbf  = (__bf16*)(ws + off); off += XN * 2;   // [B,NH,S,HD]
  __bf16* Vtbf = (__bf16*)(ws + off); off += XN * 2;   // [B,NH,HD,S] (transposed)
  __bf16* Obf  = Xbf;   // X dead after V projection; alias saves 16 MB

  const dim3 gemmGrid(M / 128, H_DIM / 128);    // (32, 16)
  const float qscale = 0.08838834764831845f;    // 1/sqrt(128)

  cast_f32_bf16<<<4096, 256, 0, stream>>>(hs, Xbf, (int)XN);
  cast_f32_bf16<<<4096, 256, 0, stream>>>(Wq, Wbf0, (int)HH);
  cast_f32_bf16<<<4096, 256, 0, stream>>>(Wk, Wbf1, (int)HH);
  cast_f32_bf16<<<4096, 256, 0, stream>>>(Wv, Wbf2, (int)HH);

  gemm_x_wT<0><<<gemmGrid, 256, 0, stream>>>(Xbf, Wbf0, bq, Qbf,
                                             (int)M, H_DIM, H_DIM, qscale);
  gemm_x_wT<0><<<gemmGrid, 256, 0, stream>>>(Xbf, Wbf1, bk, Kbf,
                                             (int)M, H_DIM, H_DIM, 1.0f);
  gemm_x_wT<2><<<gemmGrid, 256, 0, stream>>>(Xbf, Wbf2, bv, Vtbf,
                                             (int)M, H_DIM, H_DIM, 1.0f);

  flash_attn<<<dim3(SEQ / 128, BATCH * NHEADS), 256, 0, stream>>>(Qbf, Kbf, Vtbf, Obf);

  cast_f32_bf16<<<4096, 256, 0, stream>>>(Wo, Wbf0, (int)HH);
  gemm_x_wT<1><<<gemmGrid, 256, 0, stream>>>(Obf, Wbf0, bo, d_out,
                                             (int)M, H_DIM, H_DIM, 1.0f);
}
